// Attention_7172595384411
// MI455X (gfx1250) — compile-verified, matrix operand order checked by reading, not run on hardware
//
#include <hip/hip_runtime.h>

// ---------------------------------------------------------------------------
// Galerkin linear attention for MI455X (gfx1250, wave32, WMMA bf16 path).
// ---------------------------------------------------------------------------

typedef __attribute__((ext_vector_type(16))) __bf16 bfx16;
typedef __attribute__((ext_vector_type(8)))  __bf16 bfx8;
typedef __attribute__((ext_vector_type(8)))  float  f32x8;

#define DM 128
#define NROWS 65536
#define LDS_STRIDE 136   // padded bf16 row stride: bank-conflict free, 16B-aligned
#define NBLK_A 512       // NROWS / 128

__device__ __forceinline__ f32x8 wmma_bf16(bfx16 a, bfx16 b, f32x8 c) {
  return __builtin_amdgcn_wmma_f32_16x16x32_bf16(false, a, false, b, (short)0, c,
                                                 false, false);
}

// 16 bf16 elements of a 16x32 A (or 32x16 B) fragment: two 16-byte chunks,
// second chunk +16 elements (K=+16 per the 16-bit A-matrix VGPR layout).
__device__ __forceinline__ bfx16 frag_bf16(const __bf16* p) {
  union { bfx16 v; bfx8 h[2]; } u;
  u.h[0] = *(const bfx8*)(p);
  u.h[1] = *(const bfx8*)(p + 16);
  return u.v;
}

__device__ __forceinline__ bfx16 frag_lds(const __bf16* base, int c, int k0, int hi) {
  return frag_bf16(base + c * LDS_STRIDE + k0 + hi * 8);
}

// Build an A fragment from f32 global rows, converting to bf16.
__device__ __forceinline__ bfx16 frag_x_f32(const float* __restrict__ X, int row,
                                            int kb, int hi) {
  const float* p = X + (size_t)row * DM + kb + hi * 8;
  float4 f0 = *(const float4*)(p);
  float4 f1 = *(const float4*)(p + 4);
  float4 f2 = *(const float4*)(p + 16);
  float4 f3 = *(const float4*)(p + 20);
  bfx16 v;
  v[0]  = (__bf16)f0.x; v[1]  = (__bf16)f0.y; v[2]  = (__bf16)f0.z; v[3]  = (__bf16)f0.w;
  v[4]  = (__bf16)f1.x; v[5]  = (__bf16)f1.y; v[6]  = (__bf16)f1.z; v[7]  = (__bf16)f1.w;
  v[8]  = (__bf16)f2.x; v[9]  = (__bf16)f2.y; v[10] = (__bf16)f2.z; v[11] = (__bf16)f2.w;
  v[12] = (__bf16)f3.x; v[13] = (__bf16)f3.y; v[14] = (__bf16)f3.z; v[15] = (__bf16)f3.w;
  return v;
}

__device__ __forceinline__ void load_a4(bfx16 a[4], const float* __restrict__ X,
                                        int row, int hi) {
#pragma unroll
  for (int kb = 0; kb < 4; ++kb) a[kb] = frag_x_f32(X, row, kb * 32, hi);
}

// One 16x16 output tile of X @ W^T + b: K-loop of 4 WMMAs, B from LDS-staged W.
__device__ __forceinline__ f32x8 proj_tile(const bfx16 a[4], const __bf16* s_w,
                                           int nt, int lrow, int hi, float bias) {
  f32x8 acc;
#pragma unroll
  for (int r = 0; r < 8; ++r) acc[r] = bias;
#pragma unroll
  for (int kb = 0; kb < 4; ++kb) {
    bfx16 b = frag_lds(s_w, nt * 16 + lrow, kb * 32, hi);
    acc = wmma_bf16(a[kb], b, acc);
  }
  return acc;
}

// ---------------------------------------------------------------------------
// Kernel A: projections Q,K,V (bf16 WMMA, f32 accum) + per-block K^T V partial
// ---------------------------------------------------------------------------
__global__ __launch_bounds__(256) void galerkin_proj_kv(
    const float* __restrict__ Xq, const float* __restrict__ Xk,
    const float* __restrict__ Xv,
    const float* __restrict__ Wq, const float* __restrict__ Bq,
    const float* __restrict__ Wk, const float* __restrict__ Bk,
    const float* __restrict__ Wv, const float* __restrict__ Bv,
    __bf16* __restrict__ qbuf, float* __restrict__ partials) {
  __shared__ __align__(16) __bf16 s_w [DM * LDS_STRIDE];  // staged weights (bf16)
  __shared__ __align__(16) __bf16 s_kt[DM * LDS_STRIDE];  // K tile, transposed
  __shared__ __align__(16) __bf16 s_vt[DM * LDS_STRIDE];  // V tile, transposed

  const int tid  = threadIdx.x;
  const int l    = tid & 31;     // wave32 lane
  const int w    = tid >> 5;     // wave id (8 waves)
  const int lrow = l & 15;
  const int hi   = l >> 4;
  const int blk  = blockIdx.x;
  const int rowg = blk * 128 + w * 16;

  bfx16 a[4];

  // ---------------- Q = Xq @ Wq^T + bq  -> qbuf (bf16, row major) ----------
  for (int i = tid; i < DM * DM; i += 256)
    s_w[(i >> 7) * LDS_STRIDE + (i & 127)] = (__bf16)Wq[i];
  __syncthreads();
  load_a4(a, Xq, rowg + lrow, hi);
#pragma unroll
  for (int nt = 0; nt < 8; ++nt) {
    f32x8 acc = proj_tile(a, s_w, nt, lrow, hi, Bq[nt * 16 + lrow]);
#pragma unroll
    for (int r = 0; r < 8; ++r) {
      int m = r + hi * 8;
      qbuf[(size_t)(rowg + m) * DM + nt * 16 + lrow] = (__bf16)acc[r];
    }
  }
  __syncthreads();

  // ---------------- K = Xk @ Wk^T + bk  -> s_kt (transposed) ---------------
  for (int i = tid; i < DM * DM; i += 256)
    s_w[(i >> 7) * LDS_STRIDE + (i & 127)] = (__bf16)Wk[i];
  __syncthreads();
  load_a4(a, Xk, rowg + lrow, hi);
#pragma unroll
  for (int nt = 0; nt < 8; ++nt) {
    f32x8 acc = proj_tile(a, s_w, nt, lrow, hi, Bk[nt * 16 + lrow]);
#pragma unroll
    for (int r = 0; r < 8; ++r)
      s_kt[(nt * 16 + lrow) * LDS_STRIDE + (w * 16 + r + hi * 8)] = (__bf16)acc[r];
  }
  __syncthreads();

  // ---------------- V = Xv @ Wv^T + bv  -> s_vt (transposed) ---------------
  for (int i = tid; i < DM * DM; i += 256)
    s_w[(i >> 7) * LDS_STRIDE + (i & 127)] = (__bf16)Wv[i];
  __syncthreads();
  load_a4(a, Xv, rowg + lrow, hi);
#pragma unroll
  for (int nt = 0; nt < 8; ++nt) {
    f32x8 acc = proj_tile(a, s_w, nt, lrow, hi, Bv[nt * 16 + lrow]);
#pragma unroll
    for (int r = 0; r < 8; ++r)
      s_vt[(nt * 16 + lrow) * LDS_STRIDE + (w * 16 + r + hi * 8)] = (__bf16)acc[r];
  }
  __syncthreads();

  // ------------- partial K_h^T V_h over this block's 128 rows --------------
  // 16 output tiles (4 heads x 2 dtiles x 2 etiles), 2 per wave, K-loop 4.
#pragma unroll
  for (int tt = 0; tt < 2; ++tt) {
    int t  = w * 2 + tt;
    int h  = t >> 2;
    int dt = (t >> 1) & 1;
    int et = t & 1;
    f32x8 acc;
#pragma unroll
    for (int r = 0; r < 8; ++r) acc[r] = 0.0f;
#pragma unroll
    for (int kb = 0; kb < 4; ++kb) {
      bfx16 ka = frag_lds(s_kt, h * 32 + dt * 16 + lrow, kb * 32, hi);  // A = K^T
      bfx16 vb = frag_lds(s_vt, h * 32 + et * 16 + lrow, kb * 32, hi);  // B = V
      acc = wmma_bf16(ka, vb, acc);
    }
    float* P = partials + (size_t)blk * 4096 + h * 1024;
#pragma unroll
    for (int r = 0; r < 8; ++r) {
      int d = dt * 16 + r + hi * 8;
      P[d * 32 + et * 16 + lrow] = acc[r];
    }
  }
}

// ---------------------------------------------------------------------------
// Kernel B: reduce 512 partials -> p_attn (f32 to d_out tail, bf16^T to ws)
// ---------------------------------------------------------------------------
__global__ __launch_bounds__(256) void galerkin_reduce(
    const float* __restrict__ partials, float* __restrict__ pattn_out,
    __bf16* __restrict__ pT) {
  int idx = blockIdx.x * 256 + threadIdx.x;  // < 4096 : [h][d][e]
  float s = 0.0f;
  for (int b = 0; b < NBLK_A; ++b) s += partials[(size_t)b * 4096 + idx];
  float val = s * (1.0f / 65536.0f);
  pattn_out[idx] = val;
  int h = idx >> 10, d = (idx >> 5) & 31, e = idx & 31;
  pT[h * 1024 + e * 32 + d] = (__bf16)val;  // transposed for B-fragment reads
}

// ---------------------------------------------------------------------------
// Kernel C: x = Q @ p_attn, write (h,d,n)-flat output via LDS transpose
// ---------------------------------------------------------------------------
__global__ __launch_bounds__(256) void galerkin_qp(
    const __bf16* __restrict__ qbuf, const __bf16* __restrict__ pT,
    float* __restrict__ out) {
  __shared__ __align__(16) float s_x[128 * 129];
  const int tid  = threadIdx.x;
  const int l    = tid & 31;
  const int w    = tid >> 5;
  const int lrow = l & 15;
  const int hi   = l >> 4;
  const int n0   = blockIdx.x * 128;
  const int row  = n0 + w * 16 + lrow;

#pragma unroll
  for (int h = 0; h < 4; ++h) {
    bfx16 aq = frag_bf16(qbuf + (size_t)row * DM + h * 32 + hi * 8);
#pragma unroll
    for (int et = 0; et < 2; ++et) {
      bfx16 bp = frag_bf16(pT + h * 1024 + (et * 16 + lrow) * 32 + hi * 8);
      f32x8 acc;
#pragma unroll
      for (int r = 0; r < 8; ++r) acc[r] = 0.0f;
      acc = wmma_bf16(aq, bp, acc);
#pragma unroll
      for (int r = 0; r < 8; ++r)
        s_x[(w * 16 + r + hi * 8) * 129 + h * 32 + et * 16 + lrow] = acc[r];
    }
  }
  __syncthreads();
  // coalesced transposed write: out[(h*32+d)*65536 + n]
#pragma unroll 4
  for (int idx = tid; idx < 128 * 128; idx += 256) {
    int c = idx >> 7, r = idx & 127;
    out[(size_t)c * NROWS + n0 + r] = s_x[r * 129 + c];
  }
}

// ---------------------------------------------------------------------------
extern "C" void kernel_launch(void* const* d_in, const int* in_sizes, int n_in,
                              void* d_out, int out_size, void* d_ws, size_t ws_size,
                              hipStream_t stream) {
  (void)in_sizes; (void)n_in; (void)out_size; (void)ws_size;
  const float* q  = (const float*)d_in[0];
  const float* k  = (const float*)d_in[1];
  const float* v  = (const float*)d_in[2];
  const float* Wq = (const float*)d_in[3];
  const float* bq = (const float*)d_in[4];
  const float* Wk = (const float*)d_in[5];
  const float* bk = (const float*)d_in[6];
  const float* Wv = (const float*)d_in[7];
  const float* bv = (const float*)d_in[8];
  float* out = (float*)d_out;

  char* ws = (char*)d_ws;
  __bf16* qbuf    = (__bf16*)(ws);                            // 16,777,216 B
  float*  part    = (float*)(ws + (size_t)16777216);          //  8,388,608 B
  __bf16* pT      = (__bf16*)(ws + (size_t)16777216 + 8388608); //    8,192 B

  galerkin_proj_kv<<<NBLK_A, 256, 0, stream>>>(q, k, v, Wq, bq, Wk, bk, Wv, bv,
                                               qbuf, part);
  galerkin_reduce<<<16, 256, 0, stream>>>(part, out + (size_t)NROWS * DM, pT);
  galerkin_qp<<<NBLK_A, 256, 0, stream>>>(qbuf, pT, out);
}
